// Encoder_72713796321419
// MI455X (gfx1250) — matrix operand order repaired and checked by reading, not compiled, not run on hardware
//
#include <hip/hip_runtime.h>

typedef __attribute__((ext_vector_type(16))) _Float16 v16h;
typedef __attribute__((ext_vector_type(8)))  float    v8f;

// Problem constants (match reference)
constexpr int NB   = 2;     // batch
constexpr int NT   = 10;    // time steps
constexpr int NHC  = 32;    // hidden channels
constexpr int NGC  = 128;   // gate channels = 4*NHC
constexpr int NS   = 32;    // output spatial extent
constexpr int NSI  = 64;    // input spatial extent
constexpr int NKB  = 28;    // 27 Wh tap k-blocks + 1 padded Wx k-block
constexpr int HSROW = 34;   // staged h row: w' in [-1,32]
constexpr int XSROW = 66;   // staged x row: w' in [-1,64] (65, padded to 66)

union V16H { v16h v; _Float16 e[16]; };

// ---------------------------------------------------------------------------
// Pack Wh (128,32,3,3,3) and Wx (128,1,3,3,3) f32 weights into f16 A-fragments
// laid out in the contiguous K order this code uses for both A and B:
//   lanes 0-15 : e[0..15] = K 0..15   (M = lane)
//   lanes 16-31: e[0..15] = K 16..31  (M = lane-16)
// Blocks kb=0..26: K = input channel ic, tap = kb. Block kb=27: K = tap (<27).
// Layout: Apack[kb][mtile(8)][lane(32)][16 halves]
// ---------------------------------------------------------------------------
__global__ void pack_weights(const float* __restrict__ Wx,
                             const float* __restrict__ Wh,
                             _Float16* __restrict__ Apack) {
  int g    = blockIdx.x * blockDim.x + threadIdx.x;   // 28*8*32 = 7168 threads
  int kb   = g >> 8;
  int rem  = g & 255;
  int mt   = rem >> 5;
  int lane = rem & 31;
  int M    = mt * 16 + (lane & 15);
  _Float16* dst = Apack + ((size_t)(kb * 8 + mt) * 32 + lane) * 16;
#pragma unroll
  for (int i = 0; i < 16; ++i) {
    int K;
    if (lane < 16) K = i;          // 0..15
    else           K = i + 16;     // 16..31
    float v = 0.f;
    if (kb < 27)       v = Wh[((size_t)M * NHC + K) * 27 + kb];
    else if (K < 27)   v = Wx[(size_t)M * 27 + K];
    dst[i] = (_Float16)v;
  }
}

__device__ __forceinline__ float sigm(float x) {
  return 1.f / (1.f + __expf(-x));
}

// ---------------------------------------------------------------------------
// One ConvLSTM time step. Workgroup = 256 threads (8 waves), owns one
// (batch, d, h) z-row of 32 output voxels and all 128 gate channels.
// Implicit GEMM: M=128 (oc), N=32 (voxels), K=28 blocks of 32.
// ---------------------------------------------------------------------------
__global__ __launch_bounds__(256)
void lstm_step(const float* __restrict__ xall,   // (B,T,1,64,64,64)
               const float* __restrict__ bias,   // (128,)
               const _Float16* __restrict__ Apack,
               const float* __restrict__ h_in,   // (B,32,32,32,32)
               const float* __restrict__ c_in,
               float* __restrict__ h_out,
               float* __restrict__ c_out,
               int t) {
  __shared__ _Float16 hs[9 * HSROW * NHC];   // staged h halo: [row9][wi34][ic32]
  __shared__ _Float16 xs[9 * XSROW];         // staged x halo: [row9][wi66]
  __shared__ float    gates[NGC * NS];       // [oc128][n32]

  const int b   = blockIdx.z;
  const int d   = blockIdx.y;
  const int h   = blockIdx.x;
  const int tid = threadIdx.x;

  // ---- stage h_in halo tile (zero-padded), f32 -> f16, ic contiguous
  for (int idx = tid; idx < 9 * HSROW * NHC; idx += 256) {
    int ic = idx & 31;
    int wi = (idx >> 5) % HSROW;
    int r  = idx / (HSROW * NHC);
    int dd = d + r / 3 - 1;
    int hh = h + r % 3 - 1;
    int ww = wi - 1;
    float v = 0.f;
    if ((unsigned)dd < (unsigned)NS && (unsigned)hh < (unsigned)NS &&
        (unsigned)ww < (unsigned)NS)
      v = h_in[(((size_t)(b * NHC + ic) * NS + dd) * NS + hh) * NS + ww];
    hs[idx] = (_Float16)v;
  }
  // ---- stage x_t halo tile (stride-2 footprint, zero-padded)
  const float* xt = xall + ((size_t)b * NT + t) * NSI * NSI * NSI;
  for (int idx = tid; idx < 9 * XSROW; idx += 256) {
    int wi = idx % XSROW;
    int r  = idx / XSROW;
    int dd = 2 * d + r / 3 - 1;
    int hh = 2 * h + r % 3 - 1;
    int ww = wi - 1;
    float v = 0.f;
    if ((unsigned)dd < (unsigned)NSI && (unsigned)hh < (unsigned)NSI &&
        (unsigned)ww < (unsigned)NSI)
      v = xt[((size_t)dd * NSI + hh) * NSI + ww];
    xs[idx] = (_Float16)v;
  }
  __syncthreads();

  const int wave = tid >> 5;            // oc-tile 0..7
  const int lane = tid & 31;
  const int N    = lane & 15;
  const int klo  = (lane < 16) ? 0 : 16;  // B fragment: lane holds contiguous K run

  v8f acc0 = {};   // N = 0..15
  v8f acc1 = {};   // N = 16..31

  for (int kb = 0; kb < NKB; ++kb) {
    // A fragment: one 32-byte pre-packed load per lane
    v16h a = *(const v16h*)(Apack + ((size_t)(kb * 8 + wave) * 32 + lane) * 16);
    v16h b0, b1;
    if (kb < 27) {
      // Wh tap block: K = ic (contiguous in LDS) at fixed neighbor position
      int kd = kb / 9, kh = (kb / 3) % 3, kw = kb % 3;
      int row = kd * 3 + kh;
      const _Float16* base = &hs[row * HSROW * NHC + klo];
      b0 = *(const v16h*)(base + (size_t)(N + kw) * NHC);        // ds_load 32B
      b1 = *(const v16h*)(base + (size_t)(16 + N + kw) * NHC);
    } else {
      // Wx block: K = tap index (0..26), gather from x stage
      V16H u0, u1;
#pragma unroll
      for (int i = 0; i < 16; ++i) {
        int k = klo + i;
        _Float16 v0 = (_Float16)0.f, v1 = (_Float16)0.f;
        if (k < 27) {
          int kd = k / 9, kh = (k / 3) % 3, kw = k % 3;
          int row = kd * 3 + kh;
          v0 = xs[row * XSROW + 2 * N + kw];
          v1 = xs[row * XSROW + 2 * (16 + N) + kw];
        }
        u0.e[i] = v0; u1.e[i] = v1;
      }
      b0 = u0.v; b1 = u1.v;
    }
    acc0 = __builtin_amdgcn_wmma_f32_16x16x32_f16(false, a, false, b0,
                                                  (short)0, acc0, false, false);
    acc1 = __builtin_amdgcn_wmma_f32_16x16x32_f16(false, a, false, b1,
                                                  (short)0, acc1, false, false);
  }

  // ---- scatter C tiles (16x16 f32 layout: VGPR r -> M = r (+8 for hi lanes))
  {
    int rowBase = wave * 16 + ((lane < 16) ? 0 : 8);
    int col = lane & 15;
#pragma unroll
    for (int r = 0; r < 8; ++r) {
      gates[(rowBase + r) * NS + col]      = acc0[r];
      gates[(rowBase + r) * NS + 16 + col] = acc1[r];
    }
  }
  __syncthreads();

  // ---- fused LSTM cell update: i,f,o,g from LDS, c/h to global
  for (int e = tid; e < NHC * NS; e += 256) {
    int hc = e >> 5;
    int n  = e & 31;
    float gi = sigm(gates[(hc)      * NS + n] + bias[hc]);
    float gf = sigm(gates[(32 + hc) * NS + n] + bias[32 + hc]);
    float go = sigm(gates[(64 + hc) * NS + n] + bias[64 + hc]);
    float gg = tanhf(gates[(96 + hc) * NS + n] + bias[96 + hc]);
    size_t off = (((size_t)(b * NHC + hc) * NS + d) * NS + h) * NS + n;
    float cn = gf * c_in[off] + gi * gg;
    c_out[off] = cn;
    h_out[off] = go * tanhf(cn);
  }
}

// ---------------------------------------------------------------------------
extern "C" void kernel_launch(void* const* d_in, const int* in_sizes, int n_in,
                              void* d_out, int out_size, void* d_ws, size_t ws_size,
                              hipStream_t stream) {
  const float* x    = (const float*)d_in[0];   // (2,10,1,64,64,64)
  const float* Wx   = (const float*)d_in[1];   // (128,1,3,3,3)
  const float* Wh   = (const float*)d_in[2];   // (128,32,3,3,3)
  const float* bias = (const float*)d_in[3];   // (128,)

  const size_t HSZ = (size_t)NB * NHC * NS * NS * NS;   // 2,097,152 floats
  float* h_ping = (float*)d_ws;
  float* c_ping = h_ping + HSZ;
  float* h_pong = c_ping + HSZ;
  float* c_pong = h_pong + HSZ;
  _Float16* Apack = (_Float16*)(c_pong + HSZ);          // ws offset 32 MiB

  // h0 = c0 = 0 (graph-capture-legal stream memset)
  (void)hipMemsetAsync(d_ws, 0, 2 * HSZ * sizeof(float), stream);

  pack_weights<<<NKB, 256, 0, stream>>>(Wx, Wh, Apack);

  dim3 grid(NS, NS, NB);   // (h, d, b); each block = one z-row of 32 voxels
  for (int t = 0; t < NT; ++t) {
    const float* h_in = (t & 1) ? h_pong : h_ping;
    const float* c_in = (t & 1) ? c_pong : c_ping;
    float* h_o = (t == NT - 1) ? (float*)d_out
                               : ((t & 1) ? h_ping : h_pong);
    float* c_o = (t & 1) ? c_ping : c_pong;
    lstm_step<<<grid, 256, 0, stream>>>(x, bias, Apack, h_in, c_in, h_o, c_o, t);
  }
}
